// Block_12463995093348
// MI455X (gfx1250) — matrix operand order repaired and checked by reading, not compiled, not run on hardware
//
#include <hip/hip_runtime.h>

// ---------------------------------------------------------------------------
// Types for CDNA5 WMMA (gfx1250, wave32)
// ---------------------------------------------------------------------------
typedef __attribute__((ext_vector_type(8)))  __bf16 v8bf;
typedef __attribute__((ext_vector_type(16))) __bf16 v16bf;
typedef __attribute__((ext_vector_type(8)))  float  v8f;

#define HH    64
#define WWID  64
#define CCH   256
#define NHEAD 8
#define NPTS  4
#define DHEAD 32
#define NTOK  32768   // B*N = 2*16384
#define NPIX  8192    // B*H*W = 2*64*64
#define MLPH  1024

__device__ __forceinline__ unsigned short f2bf(float f) {
  unsigned int u = __builtin_bit_cast(unsigned int, f);
  u += 0x7FFFu + ((u >> 16) & 1u);           // round-to-nearest-even
  return (unsigned short)(u >> 16);
}

__device__ __forceinline__ float gelu_tanh(float x) {
  const float c0 = 0.7978845608028654f;      // sqrt(2/pi)
  const float c1 = 0.044715f;
  float t = tanhf(c0 * (x + c1 * x * x * x));
  return 0.5f * x * (1.0f + t);
}

// gfx1250 async global->LDS copy of 16 bytes (one lane), tracked by ASYNCcnt.
__device__ __forceinline__ void async_copy_b128(unsigned lds_addr, const void* gptr) {
  asm volatile("global_load_async_to_lds_b128 %0, %1, off"
               :: "v"(lds_addr), "v"((unsigned long long)(uintptr_t)gptr)
               : "memory");
}
__device__ __forceinline__ void wait_asynccnt0() {
  asm volatile("s_wait_asynccnt 0x0" ::: "memory");
}

// ---------------------------------------------------------------------------
// Elementwise f32 -> bf16 cast (cameras / A-side activations)
// ---------------------------------------------------------------------------
__global__ __launch_bounds__(256) void cast_bf16_kernel(
    const float* __restrict__ X, unsigned short* __restrict__ Y, int n) {
  int i = blockIdx.x * 256 + threadIdx.x;
  if (i < n) Y[i] = f2bf(X[i]);
}

// Transpose-cast weights: W[K,N] f32 -> Wt[N,K] bf16 (one-time, tiny)
__global__ __launch_bounds__(256) void castT_bf16_kernel(
    const float* __restrict__ W, unsigned short* __restrict__ Wt, int K, int N) {
  int i = blockIdx.x * 256 + threadIdx.x;
  if (i < K * N) {
    int n = i / K, k = i % K;                   // coalesced writes into Wt
    Wt[i] = f2bf(W[(size_t)k * N + n]);
  }
}

// Pack W_off[256,64] | W_w[256,32] -> transposed bf16 [96,256]; biases -> f32 [96]
__global__ __launch_bounds__(256) void pack_offwT_kernel(
    const float* __restrict__ Woff, const float* __restrict__ Ww,
    const float* __restrict__ boff, const float* __restrict__ bw,
    unsigned short* __restrict__ Wt, float* __restrict__ bc) {
  int i = blockIdx.x * 256 + threadIdx.x;
  if (i < 96 * 256) {
    int n = i / 256, k = i % 256;
    float v = (n < 64) ? Woff[(size_t)k * 64 + n] : Ww[(size_t)k * 32 + (n - 64)];
    Wt[i] = f2bf(v);
  }
  if (i < 96) bc[i] = (i < 64) ? boff[i] : bw[i - 64];
}

// ---------------------------------------------------------------------------
// Row LayerNorm over C=256, fp32 in -> bf16 out.  One wave32 per row.
// ---------------------------------------------------------------------------
__global__ __launch_bounds__(256) void layernorm_bf16_kernel(
    const float* __restrict__ X, const float* __restrict__ g,
    const float* __restrict__ b, unsigned short* __restrict__ Y, int rows) {
  const int wave = threadIdx.x >> 5, lane = threadIdx.x & 31;
  const int row = blockIdx.x * 8 + wave;
  if (row >= rows) return;
  const float* xr = X + (size_t)row * CCH;
  float v[8], s = 0.f, ss = 0.f;
#pragma unroll
  for (int i = 0; i < 8; ++i) {
    v[i] = xr[lane + 32 * i];
    s += v[i]; ss += v[i] * v[i];
  }
#pragma unroll
  for (int o = 16; o > 0; o >>= 1) {
    s  += __shfl_xor(s,  o, 32);
    ss += __shfl_xor(ss, o, 32);
  }
  const float mean = s * (1.0f / 256.0f);
  const float var  = ss * (1.0f / 256.0f) - mean * mean;
  const float rstd = rsqrtf(var + 1e-5f);
#pragma unroll
  for (int i = 0; i < 8; ++i) {
    int c = lane + 32 * i;
    Y[(size_t)row * CCH + c] = f2bf((v[i] - mean) * rstd * g[c] + b[c]);
  }
}

// ---------------------------------------------------------------------------
// bf16 WMMA GEMM:  C[M,N] = A[M,K] @ Bt[N,K]^T  (+bias) (+resid)
//   A row-major bf16, Bt = B transposed (N-major) bf16.
//   fp32 accumulation via V_WMMA_F32_16X16X32_BF16 (wave32).
//   Block: 256 threads = 8 waves; tile BM=128, BN=WN*16, BK=32.
//   Double-buffered LDS, filled with GLOBAL_LOAD_ASYNC_TO_LDS_B128 (ASYNCcnt).
//   EP: 0 = +bias, f32 | 1 = +bias+resid, f32 | 2 = +bias, GELU, bf16
// ---------------------------------------------------------------------------
template <int WN, int EP>
__global__ __launch_bounds__(256) void gemm_bf16_kernel(
    const unsigned short* __restrict__ A, const unsigned short* __restrict__ Bt,
    const float* __restrict__ bias, const float* __restrict__ resid,
    void* __restrict__ Cout, int M, int N, int K) {
  constexpr int BN    = WN * 16;
  constexpr int LDSS  = 40;            // 32 + 8 ushort pad (bank stagger)
  constexpr int ATILE = 128 * LDSS;    // ushorts per A buffer
  constexpr int BTILE = BN * LDSS;     // ushorts per B buffer
  __shared__ unsigned short sA[2 * ATILE];
  __shared__ unsigned short sB[2 * BTILE];

  const int t    = threadIdx.x;
  const int wave = t >> 5;
  const int lane = t & 31;
  const int half = lane >> 4;          // half-wave select (16-bit frag layout)
  const int l16  = lane & 15;
  const int m0   = blockIdx.x * 128;
  const int n0   = blockIdx.y * BN;

  // One 16B chunk of the A / B tile (chunk index i in [0, rows*4)).
  auto stageA_chunk = [&](int buf, int k0, int i) {
    const int r = i >> 2, c = i & 3;
    const unsigned lo = (unsigned)(uintptr_t)&sA[buf * ATILE + r * LDSS + c * 8];
    async_copy_b128(lo, A + (size_t)(m0 + r) * K + k0 + c * 8);
  };
  auto stageB_chunk = [&](int buf, int k0, int i) {
    const int n = i >> 2, c = i & 3;
    const unsigned lo = (unsigned)(uintptr_t)&sB[buf * BTILE + n * LDSS + c * 8];
    async_copy_b128(lo, Bt + (size_t)(n0 + n) * K + k0 + c * 8);
  };
  // Stage one K-slice into LDS buffer `buf`; chunk counts are compile-time
  // constants so the hot loop stays branch-free (except WN=6's remainder).
  auto stage = [&](int buf, int k0) {
    stageA_chunk(buf, k0, t);          // A: 512 chunks = 2 per thread
    stageA_chunk(buf, k0, t + 256);
    constexpr int BCH = BN * 4;        // B: BN*4 chunks
#pragma unroll
    for (int ch = 0; ch < BCH / 256; ++ch) stageB_chunk(buf, k0, t + ch * 256);
    if constexpr ((BCH % 256) != 0) {
      if (t < (BCH % 256)) stageB_chunk(buf, k0, t + (BCH / 256) * 256);
    }
  };

  v8f acc[WN];
#pragma unroll
  for (int j = 0; j < WN; ++j)
#pragma unroll
    for (int r = 0; r < 8; ++r) acc[j][r] = 0.0f;

  int cur = 0;
  stage(0, 0);
  wait_asynccnt0();
  __syncthreads();

  for (int k0 = 0; k0 < K; k0 += 32) {
    // prefetch next K-slice into the other buffer while computing this one
    if (k0 + 32 < K) stage(cur ^ 1, k0 + 32);

    // ---- A fragment (16x32): half 0 -> K{0..7,16..23}; half 1 -> K{8..15,24..31}
    v16bf afrag;
    {
      const int ab = cur * ATILE + (wave * 16 + l16) * LDSS;
      const v8bf alo = *(const v8bf*)(&sA[ab + half * 8]);
      const v8bf ahi = *(const v8bf*)(&sA[ab + 16 + half * 8]);
#pragma unroll
      for (int e = 0; e < 8; ++e) { afrag[e] = alo[e]; afrag[8 + e] = ahi[e]; }
    }
    // ---- all B fragments first (batched ds reads, then back-to-back WMMA)
    v16bf bfrag[WN];
#pragma unroll
    for (int j = 0; j < WN; ++j) {
      const int nb = cur * BTILE + (j * 16 + l16) * LDSS;
      const v8bf blo = *(const v8bf*)(&sB[nb + half * 8]);
      const v8bf bhi = *(const v8bf*)(&sB[nb + 16 + half * 8]);
#pragma unroll
      for (int e = 0; e < 8; ++e) { bfrag[j][e] = blo[e]; bfrag[j][8 + e] = bhi[e]; }
    }
#pragma unroll
    for (int j = 0; j < WN; ++j) {
      acc[j] = __builtin_amdgcn_wmma_f32_16x16x32_bf16(
          false, afrag, false, bfrag[j], (short)0, acc[j], false, false);
    }

    wait_asynccnt0();       // next-tile async writes have landed
    __syncthreads();        // and everyone is done reading `cur`
    cur ^= 1;
  }

  // ---- epilogue: C/D layout -> row = half*8 + vgpr, col = l16
#pragma unroll
  for (int j = 0; j < WN; ++j) {
    const int col = n0 + j * 16 + l16;
    const float bv = bias ? bias[col] : 0.0f;
#pragma unroll
    for (int r = 0; r < 8; ++r) {
      const int row = m0 + wave * 16 + half * 8 + r;
      float v = acc[j][r] + bv;
      if (EP == 1) v += resid[(size_t)row * N + col];
      if (EP == 2) {
        ((unsigned short*)Cout)[(size_t)row * N + col] = f2bf(gelu_tanh(v));
      } else {
        ((float*)Cout)[(size_t)row * N + col] = v;
      }
    }
  }
}

// ---------------------------------------------------------------------------
// Deformable sampling: one wave32 per token, lane = DH channel.
//   offw[tok,96]: [0..63]=offsets (h*8+p*2+{x,y}), [64..95]=weight logits (h*4+p)
//   vals fp32 [B,H,W,C] (8 MB -> L2-resident); samp out bf16 [tok, 256].
// ---------------------------------------------------------------------------
__global__ __launch_bounds__(256) void sample_kernel(
    const float* __restrict__ gauss, const float* __restrict__ offw,
    const float* __restrict__ vals, unsigned short* __restrict__ samp) {
  const int wave = threadIdx.x >> 5, lane = threadIdx.x & 31;
  const int tok = blockIdx.x * 8 + wave;
  if (tok >= NTOK) return;
  const int b = tok >> 14;  // N = 16384
  const float gx = gauss[tok * 2 + 0];
  const float gy = gauss[tok * 2 + 1];
  const float* ow = offw + (size_t)tok * 96;
  const float* vb = vals + (size_t)b * (HH * WWID * CCH);

  for (int h = 0; h < NHEAD; ++h) {
    // softmax over the 4 points of this head (lane-uniform scalars)
    float l[4], e[4];
#pragma unroll
    for (int p = 0; p < 4; ++p) l[p] = ow[64 + h * 4 + p];
    float mx = fmaxf(fmaxf(l[0], l[1]), fmaxf(l[2], l[3]));
    float esum = 0.f;
#pragma unroll
    for (int p = 0; p < 4; ++p) { e[p] = expf(l[p] - mx); esum += e[p]; }
    const float einv = 1.0f / esum;

    float acc = 0.0f;
#pragma unroll
    for (int p = 0; p < 4; ++p) {
      const float wgt = e[p] * einv;
      const float lx = gx + ow[h * 8 + p * 2 + 0] * (1.0f / 64.0f);  // off / H
      const float ly = gy + ow[h * 8 + p * 2 + 1] * (1.0f / 64.0f);
      float px = fminf(fmaxf(lx * 63.0f, 0.0f), 63.0f);
      float py = fminf(fmaxf(ly * 63.0f, 0.0f), 63.0f);
      const float fx = floorf(px), fy = floorf(py);
      const int x0 = (int)fx, y0 = (int)fy;
      const int x1 = min(x0 + 1, 63), y1 = min(y0 + 1, 63);
      const float wx = px - fx, wy = py - fy;
      const float* base = vb + h * DHEAD + lane;  // channel c = h*32 + lane
      const float v00 = base[(size_t)(y0 * WWID + x0) * CCH];
      const float v01 = base[(size_t)(y0 * WWID + x1) * CCH];
      const float v10 = base[(size_t)(y1 * WWID + x0) * CCH];
      const float v11 = base[(size_t)(y1 * WWID + x1) * CCH];
      const float bil = (1.0f - wy) * ((1.0f - wx) * v00 + wx * v01)
                      +         wy  * ((1.0f - wx) * v10 + wx * v11);
      acc += wgt * bil;
    }
    samp[(size_t)tok * CCH + h * DHEAD + lane] = f2bf(acc);
  }
}

// ---------------------------------------------------------------------------
// Host-side orchestration
// ---------------------------------------------------------------------------
extern "C" void kernel_launch(void* const* d_in, const int* in_sizes, int n_in,
                              void* d_out, int out_size, void* d_ws, size_t ws_size,
                              hipStream_t stream) {
  (void)in_sizes; (void)n_in; (void)out_size; (void)ws_size;
  const float* gauss = (const float*)d_in[0];
  const float* x     = (const float*)d_in[1];   // gaussian_features [2,16384,256]
  const float* cams  = (const float*)d_in[2];   // [2,64,64,256]
  const float* g1    = (const float*)d_in[3];
  const float* b1    = (const float*)d_in[4];
  const float* Woff  = (const float*)d_in[5];
  const float* boff  = (const float*)d_in[6];
  const float* Ww    = (const float*)d_in[7];
  const float* bw    = (const float*)d_in[8];
  const float* Wv    = (const float*)d_in[9];
  const float* Wo    = (const float*)d_in[10];
  const float* bo    = (const float*)d_in[11];
  const float* g2    = (const float*)d_in[12];
  const float* b2    = (const float*)d_in[13];
  const float* Wfc1  = (const float*)d_in[14];
  const float* bfc1  = (const float*)d_in[15];
  const float* Wfc2  = (const float*)d_in[16];
  const float* bfc2  = (const float*)d_in[17];
  float* out = (float*)d_out;

  // workspace carve-out (all 256B aligned)
  char* p = (char*)d_ws;
  auto carve = [&](size_t bytes) {
    char* r = p; p += (bytes + 255) & ~(size_t)255; return r;
  };
  unsigned short* cam_bf   = (unsigned short*)carve((size_t)NPIX * CCH * 2);
  unsigned short* wv_t     = (unsigned short*)carve((size_t)CCH * CCH * 2);
  unsigned short* woffw_t  = (unsigned short*)carve((size_t)CCH * 96 * 2);
  unsigned short* wo_t     = (unsigned short*)carve((size_t)CCH * CCH * 2);
  unsigned short* wfc1_t   = (unsigned short*)carve((size_t)CCH * MLPH * 2);
  unsigned short* wfc2_t   = (unsigned short*)carve((size_t)MLPH * CCH * 2);
  float*          bias_ow  = (float*)carve(96 * 4);
  float*          vals_f   = (float*)carve((size_t)NPIX * CCH * 4);
  unsigned short* ln1_bf   = (unsigned short*)carve((size_t)NTOK * CCH * 2);
  float*          offw_f   = (float*)carve((size_t)NTOK * 96 * 4);
  unsigned short* samp_bf  = (unsigned short*)carve((size_t)NTOK * CCH * 2);
  float*          xnew_f   = (float*)carve((size_t)NTOK * CCH * 4);
  unsigned short* ln2_bf   = (unsigned short*)carve((size_t)NTOK * CCH * 2);
  unsigned short* t_bf     = (unsigned short*)carve((size_t)NTOK * MLPH * 2);

  const int camN = NPIX * CCH;
  cast_bf16_kernel<<<(camN + 255) / 256, 256, 0, stream>>>(cams, cam_bf, camN);
  castT_bf16_kernel<<<(CCH * CCH + 255) / 256, 256, 0, stream>>>(Wv, wv_t, CCH, CCH);
  castT_bf16_kernel<<<(CCH * CCH + 255) / 256, 256, 0, stream>>>(Wo, wo_t, CCH, CCH);
  castT_bf16_kernel<<<(CCH * MLPH + 255) / 256, 256, 0, stream>>>(Wfc1, wfc1_t, CCH, MLPH);
  castT_bf16_kernel<<<(MLPH * CCH + 255) / 256, 256, 0, stream>>>(Wfc2, wfc2_t, MLPH, CCH);
  pack_offwT_kernel<<<(96 * 256 + 255) / 256, 256, 0, stream>>>(
      Woff, Ww, boff, bw, woffw_t, bias_ow);

  // ln1 = LN(x) -> bf16
  layernorm_bf16_kernel<<<NTOK / 8, 256, 0, stream>>>(x, g1, b1, ln1_bf, NTOK);

  // vals = cameras @ W_v  (8192 x 256 x 256) -> f32
  gemm_bf16_kernel<4, 0><<<dim3(NPIX / 128, CCH / 64), 256, 0, stream>>>(
      cam_bf, wv_t, nullptr, nullptr, vals_f, NPIX, CCH, CCH);

  // offw = ln1 @ [W_off|W_w] + [b_off|b_w]  (32768 x 96 x 256) -> f32
  gemm_bf16_kernel<6, 0><<<dim3(NTOK / 128, 1), 256, 0, stream>>>(
      ln1_bf, woffw_t, bias_ow, nullptr, offw_f, NTOK, 96, CCH);

  // deformable bilinear sampling -> samp bf16
  sample_kernel<<<NTOK / 8, 256, 0, stream>>>(gauss, offw_f, vals_f, samp_bf);

  // xnew = x + samp @ W_o + b_o  (32768 x 256 x 256) -> f32
  gemm_bf16_kernel<4, 1><<<dim3(NTOK / 128, CCH / 64), 256, 0, stream>>>(
      samp_bf, wo_t, bo, x, xnew_f, NTOK, CCH, CCH);

  // ln2 = LN(xnew) -> bf16
  layernorm_bf16_kernel<<<NTOK / 8, 256, 0, stream>>>(xnew_f, g2, b2, ln2_bf, NTOK);

  // t = gelu(ln2 @ W_fc1 + b_fc1)  (32768 x 1024 x 256) -> bf16
  gemm_bf16_kernel<4, 2><<<dim3(NTOK / 128, MLPH / 64), 256, 0, stream>>>(
      ln2_bf, wfc1_t, bfc1, nullptr, t_bf, NTOK, MLPH, CCH);

  // out = xnew + t @ W_fc2 + b_fc2  (32768 x 256 x 1024) -> f32
  gemm_bf16_kernel<4, 1><<<dim3(NTOK / 128, CCH / 64), 256, 0, stream>>>(
      t_bf, wfc2_t, bfc2, xnew_f, out, NTOK, CCH, MLPH);
}
